// DogeCDMoE_75634374083058
// MI455X (gfx1250) — matrix operand order repaired and checked by reading, not compiled, not run on hardware
//
#include <hip/hip_runtime.h>
#include <hip/hip_bf16.h>

// ---------------------------------------------------------------------------
// Problem constants (from reference): b=1, t=2048, d=1024, inter=2048,
// heads=4, ret=64 (n=32 per codebook), NUM_KEYS=128, TOP_K=8, E=16384.
// ---------------------------------------------------------------------------
#define T_TOK   2048
#define DMODEL  1024
#define INTER   2048
#define HEADS   4
#define HALFRET 32
#define NKEYS   128
#define TOPK    8
#define NEXP    32          // HEADS * TOPK experts per token

typedef __attribute__((ext_vector_type(16))) _Float16 v16h;
typedef __attribute__((ext_vector_type(8)))  _Float16 v8h;
typedef __attribute__((ext_vector_type(8)))  float    v8f;
typedef int v4i __attribute__((vector_size(16)));    // matches async-LDS builtin param

#define HAS_ASYNC_LDS (__has_builtin(__builtin_amdgcn_global_load_async_to_lds_b128) && \
                       __has_builtin(__builtin_amdgcn_s_wait_asynccnt))

// ---------------------------------------------------------------------------
// f32 -> f16 conversion (grid-stride)
// ---------------------------------------------------------------------------
__global__ void cvt_f32_to_f16(const float* __restrict__ in,
                               _Float16* __restrict__ out, int n) {
    int i = blockIdx.x * blockDim.x + threadIdx.x;
    int stride = gridDim.x * blockDim.x;
    for (; i < n; i += stride) out[i] = (_Float16)in[i];
}

// ---------------------------------------------------------------------------
// Routing: query projection + product-key sim + two-stage top-k + softmax.
// One block per token, 128 threads.
// probs/indices layout: [t][head][k]  flat t*32 + head*8 + k
// ---------------------------------------------------------------------------
__global__ void routing_kernel(const float* __restrict__ hidden,
                               const float* __restrict__ wq,     // (256,1024)
                               const float* __restrict__ keys,   // (4,128,2,32)
                               float* __restrict__ probs,
                               int*   __restrict__ indices) {
    __shared__ float sh_h[DMODEL];
    __shared__ float sh_q[2 * HEADS * HALFRET];       // 256
    __shared__ float sh_sim[2 * HEADS * NKEYS];       // 1024: (p*4+head)*128+key
    __shared__ float sh_sx[2 * HEADS][TOPK];
    __shared__ int   sh_ix[2 * HEADS][TOPK];
    __shared__ float sh_cs[HEADS][TOPK * TOPK];
    __shared__ int   sh_ci[HEADS][TOPK * TOPK];

    const int t = blockIdx.x;
    const int tid = threadIdx.x;
    const float* hrow = hidden + (size_t)t * DMODEL;

    for (int i = tid; i < DMODEL; i += 128) sh_h[i] = hrow[i];
    __syncthreads();

    // q = hidden @ wq.T  (256 rows)
    for (int r = tid; r < 256; r += 128) {
        const float* w = wq + (size_t)r * DMODEL;
        float acc = 0.f;
        for (int j = 0; j < DMODEL; ++j) acc += sh_h[j] * w[j];
        sh_q[r] = acc;
    }
    __syncthreads();

    // sim[p][head][key] = dot(q[p,head,:], keys[head,key,p,:])
    for (int s = tid; s < 2 * HEADS * NKEYS; s += 128) {
        int p    = s >> 9;
        int head = (s >> 7) & 3;
        int key  = s & 127;
        const float* qv = sh_q + p * (HEADS * HALFRET) + head * HALFRET;
        const float* kv = keys + (size_t)(((head * NKEYS + key) * 2 + p)) * HALFRET;
        float acc = 0.f;
        for (int j = 0; j < HALFRET; ++j) acc += qv[j] * kv[j];
        sh_sim[s] = acc;
    }
    __syncthreads();

    // per-(p,head) top-8 over 128 keys (strict > keeps lowest index: jax tie-break)
    if (tid < 2 * HEADS) {
        float* simv = sh_sim + tid * NKEYS;   // tid = p*4 + head
        for (int kk = 0; kk < TOPK; ++kk) {
            float best = -3.4e38f; int bi = 0;
            for (int i = 0; i < NKEYS; ++i) {
                float v = simv[i];
                if (v > best) { best = v; bi = i; }
            }
            sh_sx[tid][kk] = best;
            sh_ix[tid][kk] = bi;
            simv[bi] = -3.4e38f;
        }
    }
    __syncthreads();

    // cartesian combine + final top-8 + softmax, one thread per head
    if (tid < HEADS) {
        const int head = tid;
        const float* sx = sh_sx[0 * HEADS + head];
        const int*   ix = sh_ix[0 * HEADS + head];
        const float* sy = sh_sx[1 * HEADS + head];
        const int*   iy = sh_ix[1 * HEADS + head];
        for (int i = 0; i < TOPK; ++i)
            for (int j = 0; j < TOPK; ++j) {
                sh_cs[head][i * TOPK + j] = sx[i] + sy[j];
                sh_ci[head][i * TOPK + j] = ix[i] * NKEYS + iy[j];
            }
        float topv[TOPK]; int topi[TOPK];
        for (int kk = 0; kk < TOPK; ++kk) {
            float best = -3.4e38f; int bi = 0;
            for (int i = 0; i < TOPK * TOPK; ++i) {
                float v = sh_cs[head][i];
                if (v > best) { best = v; bi = i; }
            }
            topv[kk] = best;
            topi[kk] = sh_ci[head][bi];
            sh_cs[head][bi] = -3.4e38f;
        }
        // softmax over the 8 selected scores (descending => max is topv[0])
        float mx = topv[0];
        float e[TOPK]; float sum = 0.f;
        for (int kk = 0; kk < TOPK; ++kk) { e[kk] = __expf(topv[kk] - mx); sum += e[kk]; }
        float inv = 1.f / sum;
        const int base = (t * HEADS + head) * TOPK;
        for (int kk = 0; kk < TOPK; ++kk) {
            probs[base + kk]   = e[kk] * inv;
            indices[base + kk] = topi[kk];
        }
    }
}

// ---------------------------------------------------------------------------
// Expert gather/compute: experts_states[t][:] = sum_e silu(h.de)*prob * ue
// One block per token, 256 threads (8 wave32s).
// Hidden row staged to LDS via gfx1250 async global->LDS copies when available.
// ---------------------------------------------------------------------------
__global__ void expert_kernel(const float* __restrict__ hidden,
                              const float* __restrict__ down,   // (16384,1024)
                              const float* __restrict__ up,     // (16384,1024)
                              const float* __restrict__ probs,
                              const int*   __restrict__ indices,
                              float* __restrict__ expstate) {
    __shared__ float sh_h[DMODEL];
    __shared__ float sh_w[NEXP];

    const int t = blockIdx.x;
    const int tid = threadIdx.x;
    const float* hrow = hidden + (size_t)t * DMODEL;

#if HAS_ASYNC_LDS
    // 1024 floats = 4KB: each of 256 lanes issues one 16B async LDS copy.
    __builtin_amdgcn_global_load_async_to_lds_b128(
        (__attribute__((address_space(1))) v4i*)(hrow + tid * 4),
        (__attribute__((address_space(3))) v4i*)(&sh_h[tid * 4]),
        0, 0);
    __builtin_amdgcn_s_wait_asynccnt(0);
#else
    for (int i = tid; i < DMODEL; i += 256) sh_h[i] = hrow[i];
#endif
    __syncthreads();

    const int wave = tid >> 5;
    const int lane = tid & 31;

    // each wave computes 4 expert gate weights
    for (int e = wave * 4; e < wave * 4 + 4; ++e) {
        const int idx = indices[t * NEXP + e];
        const float* drow = down + (size_t)idx * DMODEL;
        float acc = 0.f;
        for (int j = lane; j < DMODEL; j += 32) acc += sh_h[j] * drow[j];
        for (int off = 16; off > 0; off >>= 1) acc += __shfl_down(acc, off);
        if (lane == 0) {
            float x = acc;
            float s = x / (1.f + __expf(-x));     // silu
            sh_w[e] = s * probs[t * NEXP + e];
        }
    }
    __syncthreads();

    // each thread owns 4 output columns; accumulate weighted up_embed rows
    const int c0 = tid * 4;
    float o0 = 0.f, o1 = 0.f, o2 = 0.f, o3 = 0.f;
    for (int e = 0; e < NEXP; ++e) {
        const float w = sh_w[e];
        const int idx = indices[t * NEXP + e];
        const float* urow = up + (size_t)idx * DMODEL + c0;
        o0 += w * urow[0]; o1 += w * urow[1]; o2 += w * urow[2]; o3 += w * urow[3];
    }
    float* o = expstate + (size_t)t * DMODEL + c0;
    o[0] = o0; o[1] = o1; o[2] = o2; o[3] = o3;
}

// ---------------------------------------------------------------------------
// WMMA fragment helpers (contiguous k order used by this code for both A and B, wave32)
//   A 16x32 f16: lane m(0-15): K=0..15 contiguous
//                lane m+16   : K=16..31 contiguous
//   B 32x16 f16: lane n(0-15): K=0..15 contiguous; lane n+16: K=16..31
// Both operand matrices are row-major K-contiguous, so loads are contiguous.
// ---------------------------------------------------------------------------
__device__ __forceinline__ v16h load_afrag(const _Float16* __restrict__ base) {
    v8h lo = *(const v8h*)(base);
    v8h hi = *(const v8h*)(base + 8);
    v16h a;
#pragma unroll
    for (int i = 0; i < 8; ++i) { a[i] = lo[i]; a[8 + i] = hi[i]; }
    return a;
}

#define WMMA_F16(A_, B_, C_) \
    __builtin_amdgcn_wmma_f32_16x16x32_f16(false, (A_), false, (B_), (short)0, (C_), false, false)

// ---------------------------------------------------------------------------
// GEMM1: act16 = f16( silu(h@wg.T) * (h@wu.T) )   M=2048, N=2048, K=1024
// block = 256 threads = 8 waves arranged 2(M) x 4(N); each wave owns a 32x32
// C tile => 2 A-frags reused across 2 Bg + 2 Bu frags = 8 WMMA per K-step.
// grid = (M/64, N/128)
// ---------------------------------------------------------------------------
__global__ void gemm1_gateup(const _Float16* __restrict__ A,    // 2048x1024
                             const _Float16* __restrict__ Bg,   // 2048x1024
                             const _Float16* __restrict__ Bu,   // 2048x1024
                             _Float16* __restrict__ act) {      // 2048x2048
    constexpr int K = DMODEL, N = INTER;
    const int lane = threadIdx.x & 31;
    const int wave = threadIdx.x >> 5;
    const int m0 = blockIdx.x * 64 + (wave >> 2) * 32;   // wave's M base
    const int n0 = blockIdx.y * 128 + (wave & 3) * 32;   // wave's N base

    const int mn = lane & 15;
    const int kh = lane >> 4;
    const _Float16* pa0  = A  + (size_t)(m0 + mn) * K + kh * 16;
    const _Float16* pa1  = pa0 + (size_t)16 * K;
    const _Float16* pbg0 = Bg + (size_t)(n0 + mn) * K + kh * 16;
    const _Float16* pbg1 = pbg0 + (size_t)16 * K;
    const _Float16* pbu0 = Bu + (size_t)(n0 + mn) * K + kh * 16;
    const _Float16* pbu1 = pbu0 + (size_t)16 * K;

    v8f cg[2][2] = {}; v8f cu[2][2] = {};
    for (int k0 = 0; k0 < K; k0 += 32) {
        __builtin_prefetch(pa0 + k0 + 128, 0, 3);
        __builtin_prefetch(pa1 + k0 + 128, 0, 3);
        v16h a0  = load_afrag(pa0 + k0);
        v16h a1  = load_afrag(pa1 + k0);
        v16h bg0 = *(const v16h*)(pbg0 + k0);
        v16h bg1 = *(const v16h*)(pbg1 + k0);
        v16h bu0 = *(const v16h*)(pbu0 + k0);
        v16h bu1 = *(const v16h*)(pbu1 + k0);
        cg[0][0] = WMMA_F16(a0, bg0, cg[0][0]);
        cg[0][1] = WMMA_F16(a0, bg1, cg[0][1]);
        cg[1][0] = WMMA_F16(a1, bg0, cg[1][0]);
        cg[1][1] = WMMA_F16(a1, bg1, cg[1][1]);
        cu[0][0] = WMMA_F16(a0, bu0, cu[0][0]);
        cu[0][1] = WMMA_F16(a0, bu1, cu[0][1]);
        cu[1][0] = WMMA_F16(a1, bu0, cu[1][0]);
        cu[1][1] = WMMA_F16(a1, bu1, cu[1][1]);
    }

    const int rbase = (lane >> 4) * 8;
    const int cbase = lane & 15;
#pragma unroll
    for (int sm = 0; sm < 2; ++sm)
#pragma unroll
        for (int sn = 0; sn < 2; ++sn) {
            const int row = m0 + sm * 16 + rbase;
            const int col = n0 + sn * 16 + cbase;
#pragma unroll
            for (int r = 0; r < 8; ++r) {
                float g = cg[sm][sn][r], u = cu[sm][sn][r];
                float v = (g / (1.f + __expf(-g))) * u;     // silu(g)*u
                act[(size_t)(row + r) * N + col] = (_Float16)v;
            }
        }
}

// ---------------------------------------------------------------------------
// GEMM2: out = act @ w_down.T + experts_states   M=2048, N=1024, K=2048
// Same 32x32-per-wave tiling.  grid = (M/64, N/128)
// ---------------------------------------------------------------------------
__global__ void gemm2_down(const _Float16* __restrict__ A,      // 2048x2048
                           const _Float16* __restrict__ B,      // 1024x2048 (w_down)
                           const float* __restrict__ expstate,  // 2048x1024
                           float* __restrict__ out) {           // 2048x1024
    constexpr int K = INTER, N = DMODEL;
    const int lane = threadIdx.x & 31;
    const int wave = threadIdx.x >> 5;
    const int m0 = blockIdx.x * 64 + (wave >> 2) * 32;
    const int n0 = blockIdx.y * 128 + (wave & 3) * 32;

    const int mn = lane & 15;
    const int kh = lane >> 4;
    const _Float16* pa0 = A + (size_t)(m0 + mn) * K + kh * 16;
    const _Float16* pa1 = pa0 + (size_t)16 * K;
    const _Float16* pb0 = B + (size_t)(n0 + mn) * K + kh * 16;
    const _Float16* pb1 = pb0 + (size_t)16 * K;

    v8f c[2][2] = {};
    for (int k0 = 0; k0 < K; k0 += 32) {
        __builtin_prefetch(pa0 + k0 + 128, 0, 3);
        __builtin_prefetch(pa1 + k0 + 128, 0, 3);
        v16h a0 = load_afrag(pa0 + k0);
        v16h a1 = load_afrag(pa1 + k0);
        v16h b0 = *(const v16h*)(pb0 + k0);
        v16h b1 = *(const v16h*)(pb1 + k0);
        c[0][0] = WMMA_F16(a0, b0, c[0][0]);
        c[0][1] = WMMA_F16(a0, b1, c[0][1]);
        c[1][0] = WMMA_F16(a1, b0, c[1][0]);
        c[1][1] = WMMA_F16(a1, b1, c[1][1]);
    }

    const int rbase = (lane >> 4) * 8;
    const int cbase = lane & 15;
#pragma unroll
    for (int sm = 0; sm < 2; ++sm)
#pragma unroll
        for (int sn = 0; sn < 2; ++sn) {
            const int row = m0 + sm * 16 + rbase;
            const int col = n0 + sn * 16 + cbase;
#pragma unroll
            for (int r = 0; r < 8; ++r) {
                size_t off = (size_t)(row + r) * N + col;
                out[off] = c[sm][sn][r] + expstate[off];
            }
        }
}

// ---------------------------------------------------------------------------
// Launch
// ---------------------------------------------------------------------------
extern "C" void kernel_launch(void* const* d_in, const int* in_sizes, int n_in,
                              void* d_out, int out_size, void* d_ws, size_t ws_size,
                              hipStream_t stream) {
    const float* hidden   = (const float*)d_in[0];   // (1,2048,1024)
    const float* wq       = (const float*)d_in[1];   // (256,1024)
    const float* keys     = (const float*)d_in[2];   // (4,128,2,32)
    const float* down_emb = (const float*)d_in[3];   // (16384,1024)
    const float* up_emb   = (const float*)d_in[4];   // (16384,1024)
    const float* w_gate   = (const float*)d_in[5];   // (2048,1024)
    const float* w_up     = (const float*)d_in[6];   // (2048,1024)
    const float* w_down   = (const float*)d_in[7];   // (1024,2048)
    float* out = (float*)d_out;                      // (1,2048,1024)

    // workspace carving (all offsets 256B-aligned)
    char* ws = (char*)d_ws;
    size_t off = 0;
    float* probs    = (float*)(ws + off); off += (size_t)T_TOK * NEXP * 4;        // 256KB
    int*   indices  = (int*)  (ws + off); off += (size_t)T_TOK * NEXP * 4;        // 256KB
    _Float16* h16   = (_Float16*)(ws + off); off += (size_t)T_TOK * DMODEL * 2;   // 4MB
    _Float16* wg16  = (_Float16*)(ws + off); off += (size_t)INTER * DMODEL * 2;   // 4MB
    _Float16* wu16  = (_Float16*)(ws + off); off += (size_t)INTER * DMODEL * 2;   // 4MB
    _Float16* wd16  = (_Float16*)(ws + off); off += (size_t)DMODEL * INTER * 2;   // 4MB
    _Float16* act16 = (_Float16*)(ws + off); off += (size_t)T_TOK * INTER * 2;    // 8MB
    float* expstate = (float*)(ws + off);    off += (size_t)T_TOK * DMODEL * 4;   // 8MB

    // 1) precision conversion for the matrix-core path
    cvt_f32_to_f16<<<2048, 256, 0, stream>>>(hidden, h16,  T_TOK * DMODEL);
    cvt_f32_to_f16<<<2048, 256, 0, stream>>>(w_gate, wg16, INTER * DMODEL);
    cvt_f32_to_f16<<<2048, 256, 0, stream>>>(w_up,   wu16, INTER * DMODEL);
    cvt_f32_to_f16<<<2048, 256, 0, stream>>>(w_down, wd16, DMODEL * INTER);

    // 2) routing (top-k expert selection + softmax gates)
    routing_kernel<<<T_TOK, 128, 0, stream>>>(hidden, wq, keys, probs, indices);

    // 3) expert gather/compute -> expstate
    expert_kernel<<<T_TOK, 256, 0, stream>>>(hidden, down_emb, up_emb,
                                             probs, indices, expstate);

    // 4) dense MLP on WMMA: act = silu(h@wg.T)*(h@wu.T)
    gemm1_gateup<<<dim3(T_TOK / 64, INTER / 128), 256, 0, stream>>>(h16, wg16, wu16, act16);

    // 5) out = act @ w_down.T + expstate
    gemm2_down<<<dim3(T_TOK / 64, DMODEL / 128), 256, 0, stream>>>(act16, wd16, expstate, out);
}
